// MimiEuclideanCodebook_18983755448879
// MI455X (gfx1250) — compile-verified, matrix operand order checked by reading, not run on hardware
//
#include <hip/hip_runtime.h>

// Problem constants (from the reference): T=65536 tokens, K=2048 codes, D=256.
#define T_TOK 65536
#define K_CB  2048
#define D_DIM 256

typedef __attribute__((ext_vector_type(16))) __bf16 v16bf;
typedef __attribute__((ext_vector_type(8)))  float  v8f;

// ---------------------------------------------------------------------------
// Fragment loaders per CDNA5 ISA 7.12.2 (16-bit A 16x32 / B 32x16 layouts):
// lane l (l&15 = row/col index, l>>4 selects k-half):
//   vector elems 0..7  = K = kbase+0..7      (VGPRs 0..3, 2 elems each)
//   vector elems 8..15 = K = kbase+16..16+7  (VGPRs 4..7)
// where kbase = (l>>4)*8.  Both halves are 16B-contiguous in row-major
// [row][k] storage -> two b128 loads per fragment.
// ---------------------------------------------------------------------------

__device__ __forceinline__ v16bf load_b_frag(const __bf16* p8) {
  // p8 = &ebf[n*D + kt*32 + kbase]
  union { v16bf v; uint4 q[2]; } u;
  const uint4* p = reinterpret_cast<const uint4*>(p8);
  u.q[0] = p[0];   // k +0..7
  u.q[1] = p[2];   // k +16..23  (16 bf16 = 2 uint4 ahead)
  return u.v;
}

__device__ __forceinline__ v16bf load_a_frag(const float* p) {
  // p = &x[r*D + kt*32 + kbase], f32 source converted to bf16 on the fly
  const float4* q = reinterpret_cast<const float4*>(p);
  float4 u0 = q[0], u1 = q[1];   // k +0..7
  float4 u2 = q[4], u3 = q[5];   // k +16..23
  v16bf a;
  a[0]  = (__bf16)u0.x; a[1]  = (__bf16)u0.y; a[2]  = (__bf16)u0.z; a[3]  = (__bf16)u0.w;
  a[4]  = (__bf16)u1.x; a[5]  = (__bf16)u1.y; a[6]  = (__bf16)u1.z; a[7]  = (__bf16)u1.w;
  a[8]  = (__bf16)u2.x; a[9]  = (__bf16)u2.y; a[10] = (__bf16)u2.z; a[11] = (__bf16)u2.w;
  a[12] = (__bf16)u3.x; a[13] = (__bf16)u3.y; a[14] = (__bf16)u3.z; a[15] = (__bf16)u3.w;
  return a;
}

// ---------------------------------------------------------------------------
// Kernel 1: per codebook row, compute sum(e^2) and a bf16 copy of E.
// One wave (32 lanes) per row; lane handles 8 contiguous floats.
// ---------------------------------------------------------------------------
__global__ __launch_bounds__(256) void vq_prep_kernel(const float* __restrict__ emb,
                                                      __bf16* __restrict__ ebf,
                                                      float* __restrict__ esq) {
  const int lane = threadIdx.x & 31;
  const int row  = (blockIdx.x * blockDim.x + threadIdx.x) >> 5;
  const float* p = emb + (size_t)row * D_DIM + lane * 8;
  float4 u0 = reinterpret_cast<const float4*>(p)[0];
  float4 u1 = reinterpret_cast<const float4*>(p)[1];
  float s = u0.x*u0.x + u0.y*u0.y + u0.z*u0.z + u0.w*u0.w
          + u1.x*u1.x + u1.y*u1.y + u1.z*u1.z + u1.w*u1.w;
  union { uint4 q; __bf16 h[8]; } c;
  c.h[0] = (__bf16)u0.x; c.h[1] = (__bf16)u0.y; c.h[2] = (__bf16)u0.z; c.h[3] = (__bf16)u0.w;
  c.h[4] = (__bf16)u1.x; c.h[5] = (__bf16)u1.y; c.h[6] = (__bf16)u1.z; c.h[7] = (__bf16)u1.w;
  reinterpret_cast<uint4*>(ebf + (size_t)row * D_DIM)[lane] = c.q;
#pragma unroll
  for (int mask = 16; mask >= 1; mask >>= 1) s += __shfl_xor(s, mask, 32);
  if (lane == 0) esq[row] = s;
}

// ---------------------------------------------------------------------------
// Kernel 2: WMMA GEMM + fused argmin.
// Each wave owns 32 rows (two 16-row A blocks sharing every B fragment),
// sweeps all K=2048 codes in 16-wide tiles (ascending n, so strict '<'
// preserves first-index tie semantics), D=256 in 8 k-steps of 32.
// dist-proxy = e_sq[n] - 2*cross (the ||x||^2 term is row-constant).
// ---------------------------------------------------------------------------
__global__ __launch_bounds__(256) void vq_argmin_kernel(const float* __restrict__ x,
                                                        const __bf16* __restrict__ ebf,
                                                        const float* __restrict__ esq,
                                                        int* __restrict__ idx_i,
                                                        float* __restrict__ idx_f) {
  const int lane   = threadIdx.x & 31;
  const int gwave  = (blockIdx.x * blockDim.x + threadIdx.x) >> 5;
  const int m_base = gwave * 32;
  const int lrow   = lane & 15;
  const int kb     = (lane >> 4) << 3;   // 0 or 8

  // A fragments stay register-resident for the whole K sweep (128 VGPRs).
  v16bf a0[8], a1[8];
#pragma unroll
  for (int kt = 0; kt < 8; ++kt) {
    a0[kt] = load_a_frag(x + (size_t)(m_base +      lrow) * D_DIM + kt * 32 + kb);
    a1[kt] = load_a_frag(x + (size_t)(m_base + 16 + lrow) * D_DIM + kt * 32 + kb);
  }

  float mv0[8], mv1[8];
  int   mi0[8], mi1[8];
#pragma unroll
  for (int v = 0; v < 8; ++v) { mv0[v] = mv1[v] = 3.4e38f; mi0[v] = mi1[v] = 0; }

  const int NT = K_CB / 16;  // 128 column tiles, ascending order
  for (int nt = 0; nt < NT; ++nt) {
    const int nbase = nt * 16;
    const __bf16* eb = ebf + (size_t)(nbase + lrow) * D_DIM + kb;
    v8f acc0 = {}, acc1 = {};
#pragma unroll
    for (int kt = 0; kt < 8; ++kt) {
      v16bf b = load_b_frag(eb + kt * 32);
      acc0 = __builtin_amdgcn_wmma_f32_16x16x32_bf16(false, a0[kt], false, b,
                                                     (short)0, acc0, false, false);
      acc1 = __builtin_amdgcn_wmma_f32_16x16x32_bf16(false, a1[kt], false, b,
                                                     (short)0, acc1, false, false);
    }
    // C layout: lane holds column n = nbase + (lane&15) in every acc VGPR.
    const int   n  = nbase + lrow;
    const float es = esq[n];
#pragma unroll
    for (int v = 0; v < 8; ++v) {
      float d0 = fmaf(-2.0f, acc0[v], es);
      float d1 = fmaf(-2.0f, acc1[v], es);
      // Strict '<' with ascending n scan == first-index argmin; branchless
      // (v_cmp_lt_f32 + 2x v_cndmask), no EXEC manipulation in the hot loop.
      bool t0 = d0 < mv0[v];
      bool t1 = d1 < mv1[v];
      mv0[v] = t0 ? d0 : mv0[v];  mi0[v] = t0 ? n : mi0[v];
      mv1[v] = t1 ? d1 : mv1[v];  mi1[v] = t1 ? n : mi1[v];
    }
  }

  // Butterfly argmin across each 16-lane half; lane (l&15)==0 holds the row
  // M = v + 8*(lane>>4) result (acc VGPR v layout, ISA 7.12.2).
  // Bitwise ops keep this branchless too.
#pragma unroll
  for (int v = 0; v < 8; ++v) {
    float bv = mv0[v]; int bi = mi0[v];
#pragma unroll
    for (int mask = 8; mask >= 1; mask >>= 1) {
      float ov = __shfl_xor(bv, mask, 32);
      int   oi = __shfl_xor(bi, mask, 32);
      int tk = (int)(ov < bv) | ((int)(ov == bv) & (int)(oi < bi));
      bv = tk ? ov : bv;  bi = tk ? oi : bi;
    }
    if ((lane & 15) == 0) {
      int row = m_base + v + ((lane >> 4) << 3);
      idx_i[row] = bi;
      idx_f[row] = (float)bi;
    }
    bv = mv1[v]; bi = mi1[v];
#pragma unroll
    for (int mask = 8; mask >= 1; mask >>= 1) {
      float ov = __shfl_xor(bv, mask, 32);
      int   oi = __shfl_xor(bi, mask, 32);
      int tk = (int)(ov < bv) | ((int)(ov == bv) & (int)(oi < bi));
      bv = tk ? ov : bv;  bi = tk ? oi : bi;
    }
    if ((lane & 15) == 0) {
      int row = m_base + 16 + v + ((lane >> 4) << 3);
      idx_i[row] = bi;
      idx_f[row] = (float)bi;
    }
  }
}

// ---------------------------------------------------------------------------
// Kernel 3: decode — gather winning codebook rows (f32, float4 wide).
// ---------------------------------------------------------------------------
__global__ __launch_bounds__(256) void vq_gather_kernel(const float* __restrict__ emb,
                                                        const int* __restrict__ idx,
                                                        float* __restrict__ out) {
  const int i  = blockIdx.x * blockDim.x + threadIdx.x;  // over T * D/4
  const int t  = i >> 6;        // D/4 == 64
  const int d4 = i & 63;
  const int k  = idx[t];
  reinterpret_cast<float4*>(out)[(size_t)t * 64 + d4] =
      reinterpret_cast<const float4*>(emb)[(size_t)k * 64 + d4];
}

// ---------------------------------------------------------------------------
extern "C" void kernel_launch(void* const* d_in, const int* in_sizes, int n_in,
                              void* d_out, int out_size, void* d_ws, size_t ws_size,
                              hipStream_t stream) {
  const float* x   = (const float*)d_in[0];   // (T, D) f32
  const float* emb = (const float*)d_in[1];   // (K, D) f32

  float* q_out = (float*)d_out;                      // T*D f32 quantized
  float* idx_f = q_out + (size_t)T_TOK * D_DIM;      // T indices (as f32)

  // Workspace layout: bf16 codebook (1 MB) | e_sq (8 KB) | int indices (256 KB)
  char*   ws   = (char*)d_ws;
  __bf16* ebf  = (__bf16*)ws;
  float*  esq  = (float*)(ws + (size_t)K_CB * D_DIM * sizeof(__bf16));
  int*    idxi = (int*)(ws + (size_t)K_CB * D_DIM * sizeof(__bf16) + K_CB * sizeof(float));

  // 1) e_sq + bf16 codebook: one wave per row -> 2048 waves = 256 blocks.
  vq_prep_kernel<<<(K_CB * 32) / 256, 256, 0, stream>>>(emb, ebf, esq);

  // 2) GEMM+argmin: 2048 waves (32 rows each) -> 256 blocks of 8 waves.
  vq_argmin_kernel<<<(T_TOK / 32) * 32 / 256, 256, 0, stream>>>(x, ebf, esq, idxi, idx_f);

  // 3) Gather decode: T*(D/4) float4 threads.
  vq_gather_kernel<<<(T_TOK * (D_DIM / 4)) / 256, 256, 0, stream>>>(emb, idxi, q_out);
}
